// MultiHeadAttention_48129403519259
// MI455X (gfx1250) — matrix operand order repaired and checked by reading, not compiled, not run on hardware
//
#include <hip/hip_runtime.h>
#include <hip/hip_bf16.h>
#include <stdint.h>
#include <stddef.h>

// Problem constants (from reference): B=2, T=2048, C=1024, H=16, D=64
#define B_DIM 2
#define T_DIM 2048
#define C_DIM 1024
#define H_DIM 16
#define D_DIM 64
#define BT_DIM (B_DIM * T_DIM)   // 4096 rows
#define C3_DIM (3 * C_DIM)       // 3072

typedef __attribute__((ext_vector_type(16))) __bf16 v16bf;
typedef __attribute__((ext_vector_type(8)))  float  v8f;
typedef __attribute__((ext_vector_type(4)))  unsigned int u32x4;

union FragBF {
  v16bf bf;
  u32x4 q[2];
};

__device__ __forceinline__ unsigned short f2bf(float f) {
  unsigned int u = __float_as_uint(f);
  u += 0x7FFFu + ((u >> 16) & 1u);          // round-to-nearest-even
  return (unsigned short)(u >> 16);
}

// Async 16B global -> LDS copy (GLOBAL_LOAD_ASYNC_TO_LDS_B128, ASYNCcnt).
// VDST VGPR carries the LDS byte offset == low 32 bits of the generic
// shared-memory address (ISA 10.2).
__device__ __forceinline__ void async_copy_b128(const unsigned short* g,
                                                unsigned short* lds) {
  const unsigned lds_off = (unsigned)(uintptr_t)lds;
  const unsigned long long ga = (unsigned long long)(uintptr_t)g;
  asm volatile("global_load_async_to_lds_b128 %0, %1, off"
               :: "v"(lds_off), "v"(ga) : "memory");
}

__device__ __forceinline__ void wait_asynccnt0() {
  asm volatile("s_wait_asynccnt 0x0" ::: "memory");
}

// A-fragment (16x32 bf16, row-major source, stride ld elements).
// lane l: M = l&15; lanes 0-15 hold K = {0..7,16..23}, lanes 16-31
// hold K = {8..15,24..31}.  Two contiguous 16B loads.
__device__ __forceinline__ FragBF load_a_frag(const unsigned short* __restrict__ base,
                                              int ld, int m0, int k0) {
  const int l  = threadIdx.x & 31;
  const int m  = l & 15;
  const int kh = l >> 4;
  const unsigned short* p = base + (size_t)(m0 + m) * ld + k0 + kh * 8;
  FragBF f;
  f.q[0] = *reinterpret_cast<const u32x4*>(p);        // K = k0+kh*8 + 0..7
  f.q[1] = *reinterpret_cast<const u32x4*>(p + 16);   // K = k0+16+kh*8 + 0..7
  return f;
}

// B-fragment (32x16 bf16) loaded from an N-major (transposed) source Bt[n][k]:
// lane l: N = l&15; lanes 0-15 hold K = {0..7,16..23}, lanes 16-31 K = {8..15,24..31}.
// Works for both global and LDS sources (ds_load_b128 after inlining).
__device__ __forceinline__ FragBF load_b_fragT(const unsigned short* baseT,
                                               int ld, int n0, int k0) {
  const int l  = threadIdx.x & 31;
  const int n  = l & 15;
  const int kh = l >> 4;
  const unsigned short* p = baseT + (size_t)(n0 + n) * ld + k0 + kh * 8;
  FragBF f;
  f.q[0] = *reinterpret_cast<const u32x4*>(p);        // K = k0+kh*8 + 0..7
  f.q[1] = *reinterpret_cast<const u32x4*>(p + 16);   // K = k0+16+kh*8 + 0..7
  return f;
}

__device__ __forceinline__ v8f wmma_bf16(const FragBF& a, const FragBF& b, v8f c) {
  return __builtin_amdgcn_wmma_f32_16x16x32_bf16(false, a.bf, false, b.bf,
                                                 (short)0, c, false, false);
}

// ---------------------------------------------------------------------------
// Software-pipelined 32x64 wave-tile GEMM core: acc[mi][nj] += A @ Bt^T
// ---------------------------------------------------------------------------
__device__ __forceinline__ void gemm_core_32x64(
    const unsigned short* __restrict__ aPtr, int lda, int m0,
    const unsigned short* __restrict__ bPtr, int ldb, int n0,
    int K, v8f acc[2][4]) {
  FragBF a0 = load_a_frag(aPtr, lda, m0, 0);
  FragBF a1 = load_a_frag(aPtr, lda, m0 + 16, 0);
  FragBF b0 = load_b_fragT(bPtr, ldb, n0, 0);
  FragBF b1 = load_b_fragT(bPtr, ldb, n0 + 16, 0);
  FragBF b2 = load_b_fragT(bPtr, ldb, n0 + 32, 0);
  FragBF b3 = load_b_fragT(bPtr, ldb, n0 + 48, 0);

  for (int k0 = 0; k0 < K - 32; k0 += 32) {
    const int kn = k0 + 32;
    FragBF na0 = load_a_frag(aPtr, lda, m0, kn);
    FragBF na1 = load_a_frag(aPtr, lda, m0 + 16, kn);
    FragBF nb0 = load_b_fragT(bPtr, ldb, n0, kn);
    FragBF nb1 = load_b_fragT(bPtr, ldb, n0 + 16, kn);
    FragBF nb2 = load_b_fragT(bPtr, ldb, n0 + 32, kn);
    FragBF nb3 = load_b_fragT(bPtr, ldb, n0 + 48, kn);

    acc[0][0] = wmma_bf16(a0, b0, acc[0][0]);
    acc[1][0] = wmma_bf16(a1, b0, acc[1][0]);
    acc[0][1] = wmma_bf16(a0, b1, acc[0][1]);
    acc[1][1] = wmma_bf16(a1, b1, acc[1][1]);
    acc[0][2] = wmma_bf16(a0, b2, acc[0][2]);
    acc[1][2] = wmma_bf16(a1, b2, acc[1][2]);
    acc[0][3] = wmma_bf16(a0, b3, acc[0][3]);
    acc[1][3] = wmma_bf16(a1, b3, acc[1][3]);

    a0 = na0; a1 = na1; b0 = nb0; b1 = nb1; b2 = nb2; b3 = nb3;
  }
  acc[0][0] = wmma_bf16(a0, b0, acc[0][0]);
  acc[1][0] = wmma_bf16(a1, b0, acc[1][0]);
  acc[0][1] = wmma_bf16(a0, b1, acc[0][1]);
  acc[1][1] = wmma_bf16(a1, b1, acc[1][1]);
  acc[0][2] = wmma_bf16(a0, b2, acc[0][2]);
  acc[1][2] = wmma_bf16(a1, b2, acc[1][2]);
  acc[0][3] = wmma_bf16(a0, b3, acc[0][3]);
  acc[1][3] = wmma_bf16(a1, b3, acc[1][3]);
}

// ---------------------------------------------------------------------------
// Pre-pass kernels
// ---------------------------------------------------------------------------
__global__ void conv_f32_to_bf16(const float* __restrict__ in,
                                 unsigned short* __restrict__ out, int n) {
  int i = blockIdx.x * blockDim.x + threadIdx.x;
  if (i < n) out[i] = f2bf(in[i]);
}

// out[c][r] = bf16(in[r][c]); rows/cols multiples of 32. block (32,8).
__global__ void transpose_to_bf16(const float* __restrict__ in,
                                  unsigned short* __restrict__ out,
                                  int rows, int cols) {
  __shared__ float tile[32][33];
  const int r0 = blockIdx.y * 32;
  const int c0 = blockIdx.x * 32;
  const int tx = threadIdx.x, ty = threadIdx.y;
  #pragma unroll
  for (int i = ty; i < 32; i += 8)
    tile[i][tx] = in[(size_t)(r0 + i) * cols + c0 + tx];
  __syncthreads();
  #pragma unroll
  for (int i = ty; i < 32; i += 8)
    out[(size_t)(c0 + i) * rows + r0 + tx] = f2bf(tile[tx][i]);
}

// ---------------------------------------------------------------------------
// Kernel 1: QKV projection.  Y = Xb @ Wqkv + b ; scatter into q/k/vT (bf16).
// Wave tile: 32(M) x 64(N).  grid = (3072/64, 4096/256), block = 256 (8 waves)
// ---------------------------------------------------------------------------
__global__ __launch_bounds__(256) void qkv_gemm_kernel(
    const unsigned short* __restrict__ xb,    // [BT][C] bf16
    const unsigned short* __restrict__ wT,    // [3C][C] bf16 (N-major)
    const float* __restrict__ bias,           // [3C]
    unsigned short* __restrict__ qb,          // [B][H][T][D] (pre-scaled 1/8)
    unsigned short* __restrict__ kb,          // [B][H][T][D]
    unsigned short* __restrict__ vt) {        // [B][H][D][T]
  const int w  = threadIdx.x >> 5;
  const int l  = threadIdx.x & 31;
  const int m0 = (blockIdx.y * 8 + w) * 32;
  const int n0 = blockIdx.x * 64;

  v8f acc[2][4] = {{v8f{}, v8f{}, v8f{}, v8f{}}, {v8f{}, v8f{}, v8f{}, v8f{}}};
  gemm_core_32x64(xb, C_DIM, m0, wT, C_DIM, n0, C_DIM, acc);

  const int n = l & 15, half = l >> 4;
  #pragma unroll
  for (int j = 0; j < 4; ++j) {
    const int col   = n0 + j * 16 + n;
    const int h     = col / (3 * D_DIM);
    const int inner = col % (3 * D_DIM);
    const float bv  = bias[col];
    #pragma unroll
    for (int mi = 0; mi < 2; ++mi) {
      #pragma unroll
      for (int r = 0; r < 8; ++r) {
        const int row = m0 + mi * 16 + half * 8 + r;  // bt index
        const int bb  = row >> 11;                    // / T_DIM
        const int t   = row & (T_DIM - 1);
        const float v = acc[mi][j][r] + bv;
        if (inner < D_DIM) {
          qb[(((size_t)bb * H_DIM + h) * T_DIM + t) * D_DIM + inner] = f2bf(v * 0.125f);
        } else if (inner < 2 * D_DIM) {
          kb[(((size_t)bb * H_DIM + h) * T_DIM + t) * D_DIM + (inner - D_DIM)] = f2bf(v);
        } else {
          vt[(((size_t)bb * H_DIM + h) * D_DIM + (inner - 2 * D_DIM)) * T_DIM + t] = f2bf(v);
        }
      }
    }
  }
}

// ---------------------------------------------------------------------------
// Kernel 2: flash attention.  4 waves per block, all on the SAME (b,h):
// K/V chunks staged once per block into double-buffered LDS via
// global_load_async_to_lds_b128 (ASYNCcnt), overlapping the next chunk's DMA
// with the current chunk's WMMA + softmax.  grid = 1024, block = 128.
// ---------------------------------------------------------------------------
__global__ __launch_bounds__(128) void attn_kernel(
    const unsigned short* __restrict__ qb,    // [BH][T][D] (scaled)
    const unsigned short* __restrict__ kb,    // [BH][T][D]
    const unsigned short* __restrict__ vt,    // [BH][D][T]
    unsigned short* __restrict__ ab) {        // [B*T][C] bf16 (col = h*64+d)
  __shared__ __align__(16) unsigned short ktile[2][32 * 64];  // 2 x 4KB
  __shared__ __align__(16) unsigned short vtile[2][64 * 32];  // 2 x 4KB
  __shared__ __align__(16) unsigned short ptile[4][16 * 32];  // per-wave P stage

  const int tid  = threadIdx.x;
  const int w    = tid >> 5;
  const int l    = tid & 31;
  const int bh   = blockIdx.x >> 5;               // 32 blocks per head
  const int qt   = ((blockIdx.x & 31) << 2) | w;  // q-tile within head
  const int m    = l & 15;
  const int half = l >> 4;

  const unsigned short* qp = qb + (size_t)bh * T_DIM * D_DIM;
  const unsigned short* kp = kb + (size_t)bh * T_DIM * D_DIM;
  const unsigned short* vp = vt + (size_t)bh * D_DIM * T_DIM;

  // cooperative async staging of one 32-key chunk of K and V into LDS buf
  auto issue_chunk = [&](int kc, int buf) {
    // K chunk: rows [kc,kc+32) x 64 d = contiguous 4KB
    #pragma unroll
    for (int c = tid; c < 256; c += 128)
      async_copy_b128(kp + (size_t)kc * D_DIM + c * 8, &ktile[buf][c * 8]);
    // V chunk: 64 d-rows x 32 keys (row stride T in global, 32 in LDS)
    #pragma unroll
    for (int c = tid; c < 256; c += 128) {
      const int row = c >> 2, seg = c & 3;
      async_copy_b128(vp + (size_t)row * T_DIM + kc + seg * 8,
                      &vtile[buf][row * 32 + seg * 8]);
    }
  };

  FragBF qa0 = load_a_frag(qp, D_DIM, qt * 16, 0);
  FragBF qa1 = load_a_frag(qp, D_DIM, qt * 16, 32);

  v8f o[4] = {v8f{}, v8f{}, v8f{}, v8f{}};
  float mrow[8], lrow[8];
  #pragma unroll
  for (int r = 0; r < 8; ++r) { mrow[r] = -INFINITY; lrow[r] = 0.0f; }

  unsigned short* pt = ptile[w];

  constexpr int NCHUNK = T_DIM / 32;   // 64
  issue_chunk(0, 0);

  for (int it = 0; it < NCHUNK; ++it) {
    const int buf = it & 1;
    const int kc  = it * 32;

    wait_asynccnt0();     // this wave's slice of buf is in LDS
    __syncthreads();      // all slices landed; prev readers of buf^1 done

    if (it + 1 < NCHUNK) issue_chunk(kc + 32, buf ^ 1);
    if (kc + 64 < T_DIM) {     // speculative L2 warm-up 2 chunks ahead
      __builtin_prefetch(kp + (size_t)(kc + 64 + m) * D_DIM, 0, 0);
      __builtin_prefetch(vp + (size_t)m * T_DIM + kc + 64, 0, 0);
    }

    // ---- S = Q @ K^T for keys [kc, kc+32), K fragments from LDS ----
    v8f s0 = {}, s1 = {};
    {
      FragBF kb00 = load_b_fragT(ktile[buf], D_DIM, 0, 0);
      FragBF kb01 = load_b_fragT(ktile[buf], D_DIM, 0, 32);
      FragBF kb10 = load_b_fragT(ktile[buf], D_DIM, 16, 0);
      FragBF kb11 = load_b_fragT(ktile[buf], D_DIM, 16, 32);
      s0 = wmma_bf16(qa0, kb00, s0);
      s0 = wmma_bf16(qa1, kb01, s0);
      s1 = wmma_bf16(qa0, kb10, s1);
      s1 = wmma_bf16(qa1, kb11, s1);
    }

    // ---- online softmax (row reductions within 16-lane halves) ----
    float p0[8], p1[8];
    #pragma unroll
    for (int r = 0; r < 8; ++r) {
      float mx = fmaxf(s0[r], s1[r]);
      mx = fmaxf(mx, __shfl_xor(mx, 1, 32));
      mx = fmaxf(mx, __shfl_xor(mx, 2, 32));
      mx = fmaxf(mx, __shfl_xor(mx, 4, 32));
      mx = fmaxf(mx, __shfl_xor(mx, 8, 32));
      const float mn = fmaxf(mrow[r], mx);
      const float al = __expf(mrow[r] - mn);
      mrow[r] = mn;
      const float e0 = __expf(s0[r] - mn);
      const float e1 = __expf(s1[r] - mn);
      p0[r] = e0; p1[r] = e1;
      float rs = e0 + e1;
      rs += __shfl_xor(rs, 1, 32);
      rs += __shfl_xor(rs, 2, 32);
      rs += __shfl_xor(rs, 4, 32);
      rs += __shfl_xor(rs, 8, 32);
      lrow[r] = lrow[r] * al + rs;
      o[0][r] *= al; o[1][r] *= al; o[2][r] *= al; o[3][r] *= al;
    }

    // ---- re-layout P (D-frag -> A-frag) through per-wave LDS tile ----
    #pragma unroll
    for (int r = 0; r < 8; ++r) {
      const int row = half * 8 + r;
      pt[row * 32 + m]      = f2bf(p0[r]);
      pt[row * 32 + 16 + m] = f2bf(p1[r]);
    }
    // LDS ops from one wave complete in order (DScnt in-order): no barrier.
    FragBF pa;
    {
      const unsigned short* pp = pt + m * 32 + half * 8;
      pa.q[0] = *reinterpret_cast<const u32x4*>(pp);
      pa.q[1] = *reinterpret_cast<const u32x4*>(pp + 16);
    }

    // ---- O += P @ V, V fragments from LDS (chunk-local k0 = 0) ----
    {
      FragBF vb0 = load_b_fragT(vtile[buf], 32, 0, 0);
      FragBF vb1 = load_b_fragT(vtile[buf], 32, 16, 0);
      FragBF vb2 = load_b_fragT(vtile[buf], 32, 32, 0);
      FragBF vb3 = load_b_fragT(vtile[buf], 32, 48, 0);
      o[0] = wmma_bf16(pa, vb0, o[0]);
      o[1] = wmma_bf16(pa, vb1, o[1]);
      o[2] = wmma_bf16(pa, vb2, o[2]);
      o[3] = wmma_bf16(pa, vb3, o[3]);
    }
  }

  // ---- finalize: divide by l, store bf16 into attn matrix ----
  const int bb = bh >> 4;            // / H
  const int h  = bh & 15;
  #pragma unroll
  for (int r = 0; r < 8; ++r) {
    const float inv = 1.0f / lrow[r];
    const int t = qt * 16 + half * 8 + r;
    #pragma unroll
    for (int j = 0; j < 4; ++j) {
      ab[((size_t)(bb * T_DIM + t)) * C_DIM + h * D_DIM + j * 16 + m] =
          f2bf(o[j][r] * inv);
    }
  }
}

// ---------------------------------------------------------------------------
// Kernel 3: output projection.  out = attn @ W_out + b_out  (fp32 out)
// grid = (1024/64, 4096/256), block = 256
// ---------------------------------------------------------------------------
__global__ __launch_bounds__(256) void out_gemm_kernel(
    const unsigned short* __restrict__ ab,    // [BT][C] bf16
    const unsigned short* __restrict__ wT,    // [C][C] bf16 (N-major)
    const float* __restrict__ bias,           // [C]
    float* __restrict__ out) {                // [BT][C] f32
  const int w  = threadIdx.x >> 5;
  const int l  = threadIdx.x & 31;
  const int m0 = (blockIdx.y * 8 + w) * 32;
  const int n0 = blockIdx.x * 64;

  v8f acc[2][4] = {{v8f{}, v8f{}, v8f{}, v8f{}}, {v8f{}, v8f{}, v8f{}, v8f{}}};
  gemm_core_32x64(ab, C_DIM, m0, wT, C_DIM, n0, C_DIM, acc);

  const int n = l & 15, half = l >> 4;
  #pragma unroll
  for (int j = 0; j < 4; ++j) {
    const int col  = n0 + j * 16 + n;
    const float bv = bias[col];
    #pragma unroll
    for (int mi = 0; mi < 2; ++mi) {
      #pragma unroll
      for (int r = 0; r < 8; ++r) {
        const int row = m0 + mi * 16 + half * 8 + r;
        out[(size_t)row * C_DIM + col] = acc[mi][j][r] + bv;
      }
    }
  }
}

// ---------------------------------------------------------------------------
// Host launcher
// ---------------------------------------------------------------------------
extern "C" void kernel_launch(void* const* d_in, const int* in_sizes, int n_in,
                              void* d_out, int out_size, void* d_ws, size_t ws_size,
                              hipStream_t stream) {
  (void)in_sizes; (void)n_in; (void)out_size; (void)ws_size;
  const float* x    = (const float*)d_in[0];   // [B][T][C]
  const float* Wqkv = (const float*)d_in[1];   // [C][3C]
  const float* bqkv = (const float*)d_in[2];   // [3C]
  const float* Wout = (const float*)d_in[3];   // [C][C]
  const float* bout = (const float*)d_in[4];   // [C]
  float* out = (float*)d_out;

  // Workspace layout (bytes), all 256B-aligned:
  constexpr size_t SZ_XB  = (size_t)BT_DIM * C_DIM * 2;    // 8 MB
  constexpr size_t SZ_WQT = (size_t)C3_DIM * C_DIM * 2;    // 6 MB
  constexpr size_t SZ_WOT = (size_t)C_DIM * C_DIM * 2;     // 2 MB
  constexpr size_t SZ_HED = (size_t)B_DIM * H_DIM * T_DIM * D_DIM * 2;  // 8 MB
  char* ws = (char*)d_ws;
  size_t off = 0;
  unsigned short* xb  = (unsigned short*)(ws + off); off += SZ_XB;
  unsigned short* wqT = (unsigned short*)(ws + off); off += SZ_WQT;
  unsigned short* woT = (unsigned short*)(ws + off); off += SZ_WOT;
  unsigned short* qbp = (unsigned short*)(ws + off); off += SZ_HED;
  unsigned short* kbp = (unsigned short*)(ws + off); off += SZ_HED;
  unsigned short* vtp = (unsigned short*)(ws + off); off += SZ_HED;
  unsigned short* abp = (unsigned short*)(ws + off); off += SZ_XB;   // ~50.3 MB

  // 1) convert activations / transpose+convert weights to bf16
  const int nx = BT_DIM * C_DIM;
  conv_f32_to_bf16<<<nx / 256, 256, 0, stream>>>(x, xb, nx);
  transpose_to_bf16<<<dim3(C3_DIM / 32, C_DIM / 32), dim3(32, 8), 0, stream>>>(
      Wqkv, wqT, C_DIM, C3_DIM);
  transpose_to_bf16<<<dim3(C_DIM / 32, C_DIM / 32), dim3(32, 8), 0, stream>>>(
      Wout, woT, C_DIM, C_DIM);

  // 2) QKV projection + head scatter (q scaled by 1/sqrt(D), v transposed)
  qkv_gemm_kernel<<<dim3(C3_DIM / 64, BT_DIM / 256), 256, 0, stream>>>(
      xb, wqT, bqkv, qbp, kbp, vtp);

  // 3) flash attention: async LDS-staged K/V, 4 query tiles per block
  attn_kernel<<<(B_DIM * H_DIM * (T_DIM / 16)) / 4, 128, 0, stream>>>(
      qbp, kbp, vtp, abp);

  // 4) output projection
  out_gemm_kernel<<<dim3(C_DIM / 64, BT_DIM / 256), 256, 0, stream>>>(
      abp, woT, bout, out);
}
